// FPModule_18691697672884
// MI455X (gfx1250) — matrix operand order repaired and checked by reading, not compiled, not run on hardware
//
#include <hip/hip_runtime.h>

typedef _Float16 half_t;
typedef _Float16 v16h __attribute__((ext_vector_type(16)));
typedef _Float16 v8h  __attribute__((ext_vector_type(8)));
typedef float    v8f  __attribute__((ext_vector_type(8)));

// ---- static problem sizes (match reference) ----
constexpr int B_    = 8;
constexpr int NC    = 1024;
constexpr int NF    = 4096;
constexpr int NIN   = 256;
constexpr int NSKIP = 128;
constexpr int NOUT  = 256;
constexpr int DIM   = NIN + NSKIP;   // 384
constexpr int M     = B_ * NF;       // 32768 rows
constexpr float BN_EPS = 1e-5f;

// ---- workspace layout (bytes, all 256-aligned) ----
constexpr size_t OFF_W1T   = 0;                                   // DIM*DIM f16
constexpr size_t OFF_W2T   = OFF_W1T + (size_t)DIM * DIM * 2;     // NOUT*DIM f16
constexpr size_t OFF_STATS = OFF_W2T + (size_t)NOUT * DIM * 2;    // 2560 f32
constexpr size_t OFF_IDX   = OFF_STATS + 2560 * 4;                // M*3 int
constexpr size_t OFF_WGT   = OFF_IDX + (size_t)M * 3 * 4;         // M*3 f32
constexpr size_t OFF_H0    = OFF_WGT + (size_t)M * 3 * 4;         // M*DIM f16
constexpr size_t OFF_H1    = OFF_H0 + (size_t)M * DIM * 2;        // M*DIM f16
constexpr size_t OFF_A     = OFF_H1 + (size_t)M * DIM * 2;        // M*DIM f32 (a1, reused as a2)

// stats region float indices
constexpr int ST_SUM1 = 0, ST_SQ1 = 384, ST_SCALE1 = 768, ST_SHIFT1 = 1152;
constexpr int ST_SUM2 = 1536, ST_SQ2 = 1792, ST_SCALE2 = 2048, ST_SHIFT2 = 2304;
constexpr int ST_TOTAL = 2560;

// ---------------------------------------------------------------------------
// prep: transpose+convert weights to f16 [N,K], zero BN accumulators
// ---------------------------------------------------------------------------
__global__ __launch_bounds__(256)
void prep_kernel(const float* __restrict__ W1, const float* __restrict__ W2,
                 half_t* __restrict__ W1T, half_t* __restrict__ W2T,
                 float* __restrict__ stats)
{
    int i = blockIdx.x * 256 + threadIdx.x;
    constexpr int N1 = DIM * DIM;        // 147456
    constexpr int N2 = NOUT * DIM;       // 98304
    if (i < N1) {
        int n = i / DIM, k = i % DIM;    // W1T[n*DIM+k] = W1[k*DIM+n]
        W1T[i] = (half_t)W1[(size_t)k * DIM + n];
    } else if (i < N1 + N2) {
        int j = i - N1;
        int n = j / DIM, k = j % DIM;    // W2T[n*DIM+k] = W2[k*NOUT+n]
        W2T[j] = (half_t)W2[(size_t)k * NOUT + n];
    } else if (i < N1 + N2 + ST_TOTAL) {
        stats[i - N1 - N2] = 0.0f;
    }
}

// ---------------------------------------------------------------------------
// kNN top-3 per fine point (coarse positions staged in LDS)
// ---------------------------------------------------------------------------
__global__ __launch_bounds__(256)
void knn_kernel(const float* __restrict__ pos, const float* __restrict__ pos_skip,
                int* __restrict__ idx_out, float* __restrict__ w_out)
{
    __shared__ float cpx[NC], cpy[NC], cpz[NC];
    constexpr int CHUNKS = NF / 256;                 // 16
    int b     = blockIdx.x / CHUNKS;
    int chunk = blockIdx.x % CHUNKS;

    const float* cp = pos + (size_t)b * NC * 3;
    for (int t = threadIdx.x; t < NC * 3; t += 256) {
        float v = cp[t];
        int j = t / 3, comp = t % 3;
        if (comp == 0) cpx[j] = v; else if (comp == 1) cpy[j] = v; else cpz[j] = v;
    }
    __syncthreads();

    int f = b * NF + chunk * 256 + threadIdx.x;
    float qx = pos_skip[(size_t)f * 3 + 0];
    float qy = pos_skip[(size_t)f * 3 + 1];
    float qz = pos_skip[(size_t)f * 3 + 2];

    float d0 = 3.4e38f, d1 = 3.4e38f, d2 = 3.4e38f;
    int   i0 = 0, i1 = 0, i2 = 0;
    for (int j = 0; j < NC; ++j) {
        float dx = qx - cpx[j], dy = qy - cpy[j], dz = qz - cpz[j];
        float d = dx * dx + dy * dy + dz * dz;
        if (d < d2) {
            if (d < d1) {
                d2 = d1; i2 = i1;
                if (d < d0) { d1 = d0; i1 = i0; d0 = d; i0 = j; }
                else        { d1 = d;  i1 = j; }
            } else { d2 = d; i2 = j; }
        }
    }
    idx_out[(size_t)f * 3 + 0] = b * NC + i0;
    idx_out[(size_t)f * 3 + 1] = b * NC + i1;
    idx_out[(size_t)f * 3 + 2] = b * NC + i2;
    w_out[(size_t)f * 3 + 0] = 1.0f / fmaxf(d0, 1e-16f);
    w_out[(size_t)f * 3 + 1] = 1.0f / fmaxf(d1, 1e-16f);
    w_out[(size_t)f * 3 + 2] = 1.0f / fmaxf(d2, 1e-16f);
}

// ---------------------------------------------------------------------------
// interpolate + concat skip -> h0 (f16), one block per fine point
// ---------------------------------------------------------------------------
__global__ __launch_bounds__(128)
void interp_kernel(const float* __restrict__ x, const float* __restrict__ x_skip,
                   const int* __restrict__ idx, const float* __restrict__ w,
                   half_t* __restrict__ h0)
{
    int f = blockIdx.x;
    int t = threadIdx.x;
    int i0 = idx[(size_t)f * 3 + 0];
    int i1 = idx[(size_t)f * 3 + 1];
    int i2 = idx[(size_t)f * 3 + 2];
    float w0 = w[(size_t)f * 3 + 0];
    float w1 = w[(size_t)f * 3 + 1];
    float w2 = w[(size_t)f * 3 + 2];
    float inv = 1.0f / (w0 + w1 + w2);
    w0 *= inv; w1 *= inv; w2 *= inv;

    half_t* hr = h0 + (size_t)f * DIM;
#pragma unroll
    for (int cb = 0; cb < NIN; cb += 128) {
        int c = cb + t;
        float v = w0 * x[(size_t)i0 * NIN + c]
                + w1 * x[(size_t)i1 * NIN + c]
                + w2 * x[(size_t)i2 * NIN + c];
        hr[c] = (half_t)v;
    }
    hr[NIN + t] = (half_t)x_skip[(size_t)f * NSKIP + t];
}

// ---------------------------------------------------------------------------
// WMMA GEMM: C = relu(A[M,KDIM] * BT[N,KDIM]^T + bias), f32 out.
// One wave computes a 16 x (NGRP*16) tile group with v_wmma_f32_16x16x32_f16.
// ---------------------------------------------------------------------------
template <int NCOLS, int KDIM, int NGRP>
__global__ __launch_bounds__(256)
void wmma_gemm_relu(const half_t* __restrict__ A, const half_t* __restrict__ BT,
                    const float* __restrict__ bias, float* __restrict__ C)
{
    constexpr int NGROUPS = (NCOLS / 16) / NGRP;
    int wave = (blockIdx.x * 256 + threadIdx.x) >> 5;
    int lane = threadIdx.x & 31;
    int grp   = wave % NGROUPS;
    int mtile = wave / NGROUPS;
    int m0 = mtile * 16;
    int n0 = grp * (NGRP * 16);
    int lrow = lane & 15;
    int lhi  = lane >> 4;

    v8f acc[NGRP];
#pragma unroll
    for (int g = 0; g < NGRP; ++g) acc[g] = v8f{};

    // A fragment: lane holds row m0+lrow, contiguous K run of 16 at koffA
    const half_t* arow = A + (size_t)(m0 + lrow) * KDIM + lhi * 16;
    // B fragment: lane holds col n0+g*16+lrow, contiguous K run of 16 at koffB
    const half_t* bbase = BT + (size_t)(n0 + lrow) * KDIM + lhi * 16;

    for (int k0 = 0; k0 < KDIM; k0 += 32) {
        v8h alo = *(const v8h*)(arow + k0);
        v8h ahi = *(const v8h*)(arow + k0 + 8);
        v16h a = __builtin_shufflevector(alo, ahi,
                                         0, 1, 2, 3, 4, 5, 6, 7,
                                         8, 9, 10, 11, 12, 13, 14, 15);
#pragma unroll
        for (int g = 0; g < NGRP; ++g) {
            v16h b = *(const v16h*)(bbase + (size_t)g * 16 * KDIM + k0);
            acc[g] = __builtin_amdgcn_wmma_f32_16x16x32_f16(
                false, a, false, b, (short)0, acc[g], false, false);
        }
    }

    // epilogue: C layout — col = lane&15, row = 8*lhi + v
#pragma unroll
    for (int g = 0; g < NGRP; ++g) {
        int n = n0 + g * 16 + lrow;
        float bv = bias[n];
#pragma unroll
        for (int v = 0; v < 8; ++v) {
            int m = m0 + lhi * 8 + v;
            float val = acc[g][v] + bv;
            C[(size_t)m * NCOLS + n] = val > 0.0f ? val : 0.0f;
        }
    }
}

// ---------------------------------------------------------------------------
// per-column sum / sum-of-squares (atomic accumulation into zeroed stats)
// ---------------------------------------------------------------------------
template <int N>
__global__ __launch_bounds__(128)
void col_stats(const float* __restrict__ A, float* __restrict__ sum,
               float* __restrict__ sq)
{
    int r0 = blockIdx.x * 128;
    int t  = threadIdx.x;
#pragma unroll
    for (int cb = 0; cb < N; cb += 128) {
        int c = cb + t;
        float s = 0.0f, q = 0.0f;
        for (int r = 0; r < 128; ++r) {
            float v = A[(size_t)(r0 + r) * N + c];
            s += v;
            q += v * v;
        }
        atomicAdd(&sum[c], s);
        atomicAdd(&sq[c], q);
    }
}

__global__ __launch_bounds__(256)
void bn_finalize(const float* __restrict__ sum, const float* __restrict__ sq,
                 const float* __restrict__ gamma, const float* __restrict__ beta,
                 float* __restrict__ scale, float* __restrict__ shift, int N)
{
    int c = blockIdx.x * 256 + threadIdx.x;
    if (c < N) {
        const float invM = 1.0f / (float)M;
        float mean = sum[c] * invM;
        float var  = sq[c] * invM - mean * mean;   // biased variance
        float sc   = gamma[c] * rsqrtf(var + BN_EPS);
        scale[c] = sc;
        shift[c] = beta[c] - mean * sc;
    }
}

template <int N>
__global__ __launch_bounds__(256)
void bn_apply_f16(const float* __restrict__ A, const float* __restrict__ scale,
                  const float* __restrict__ shift, half_t* __restrict__ out)
{
    int i = blockIdx.x * 256 + threadIdx.x;
    int c = i % N;
    out[i] = (half_t)(A[i] * scale[c] + shift[c]);
}

template <int N>
__global__ __launch_bounds__(256)
void bn_apply_f32(const float* __restrict__ A, const float* __restrict__ scale,
                  const float* __restrict__ shift, float* __restrict__ out)
{
    int i = blockIdx.x * 256 + threadIdx.x;
    int c = i % N;
    out[i] = A[i] * scale[c] + shift[c];
}

// ---------------------------------------------------------------------------
// tail: copy pos_skip (f32) and batch_skip (i64 -> f32) into d_out
// ---------------------------------------------------------------------------
__global__ __launch_bounds__(256)
void tail_kernel(const float* __restrict__ pos_skip,
                 const long long* __restrict__ batch_skip,
                 float* __restrict__ out)
{
    int i = blockIdx.x * 256 + threadIdx.x;
    constexpr int NPOS = M * 3;
    if (i < NPOS) {
        out[i] = pos_skip[i];
    } else {
        out[i] = (float)batch_skip[i - NPOS];
    }
}

// ---------------------------------------------------------------------------
extern "C" void kernel_launch(void* const* d_in, const int* in_sizes, int n_in,
                              void* d_out, int out_size, void* d_ws, size_t ws_size,
                              hipStream_t stream)
{
    const float*     x          = (const float*)d_in[0];
    const float*     pos        = (const float*)d_in[1];
    const float*     x_skip     = (const float*)d_in[3];
    const float*     pos_skip   = (const float*)d_in[4];
    const long long* batch_skip = (const long long*)d_in[5];
    const float*     W1  = (const float*)d_in[6];
    const float*     b1  = (const float*)d_in[7];
    const float*     g1  = (const float*)d_in[8];
    const float*     bt1 = (const float*)d_in[9];
    const float*     W2  = (const float*)d_in[10];
    const float*     b2  = (const float*)d_in[11];
    const float*     g2  = (const float*)d_in[12];
    const float*     bt2 = (const float*)d_in[13];

    char* ws = (char*)d_ws;
    half_t* W1T   = (half_t*)(ws + OFF_W1T);
    half_t* W2T   = (half_t*)(ws + OFF_W2T);
    float*  stats = (float*)(ws + OFF_STATS);
    int*    knn_i = (int*)(ws + OFF_IDX);
    float*  knn_w = (float*)(ws + OFF_WGT);
    half_t* h0    = (half_t*)(ws + OFF_H0);
    half_t* h1    = (half_t*)(ws + OFF_H1);
    float*  a1    = (float*)(ws + OFF_A);
    float*  a2    = (float*)(ws + OFF_A);   // reuses a1 storage (a1 dead by then)

    float* out_h = (float*)d_out;           // [M, NOUT]
    float* out_tail = out_h + (size_t)M * NOUT;

    // 1) weight transpose/convert + zero BN accumulators
    {
        int total = DIM * DIM + NOUT * DIM + ST_TOTAL;
        prep_kernel<<<(total + 255) / 256, 256, 0, stream>>>(W1, W2, W1T, W2T, stats);
    }
    // 2) kNN (top-3)
    knn_kernel<<<B_ * (NF / 256), 256, 0, stream>>>(pos, pos_skip, knn_i, knn_w);
    // 3) interpolate + concat -> h0 (f16)
    interp_kernel<<<M, 128, 0, stream>>>(x, x_skip, knn_i, knn_w, h0);
    // 4) GEMM1 + bias + ReLU -> a1 (f32).  waves = (M/16)*(DIM/64) = 12288
    wmma_gemm_relu<DIM, DIM, 4><<<(M / 16) * (DIM / 64) / 8, 256, 0, stream>>>(
        h0, W1T, b1, a1);
    // 5) BN stats + finalize + apply -> h1 (f16)
    col_stats<DIM><<<M / 128, 128, 0, stream>>>(a1, stats + ST_SUM1, stats + ST_SQ1);
    bn_finalize<<<2, 256, 0, stream>>>(stats + ST_SUM1, stats + ST_SQ1, g1, bt1,
                                       stats + ST_SCALE1, stats + ST_SHIFT1, DIM);
    bn_apply_f16<DIM><<<(size_t)M * DIM / 256, 256, 0, stream>>>(
        a1, stats + ST_SCALE1, stats + ST_SHIFT1, h1);
    // 6) GEMM2 + bias + ReLU -> a2 (f32).  waves = (M/16)*(NOUT/64) = 8192
    wmma_gemm_relu<NOUT, DIM, 4><<<(M / 16) * (NOUT / 64) / 8, 256, 0, stream>>>(
        h1, W2T, b2, a2);
    // 7) BN stats + finalize + apply -> d_out (f32)
    col_stats<NOUT><<<M / 128, 128, 0, stream>>>(a2, stats + ST_SUM2, stats + ST_SQ2);
    bn_finalize<<<1, 256, 0, stream>>>(stats + ST_SUM2, stats + ST_SQ2, g2, bt2,
                                       stats + ST_SCALE2, stats + ST_SHIFT2, NOUT);
    bn_apply_f32<NOUT><<<(size_t)M * NOUT / 256, 256, 0, stream>>>(
        a2, stats + ST_SCALE2, stats + ST_SHIFT2, out_h);
    // 8) pos_skip + batch_skip passthrough
    tail_kernel<<<(M * 3 + M) / 256, 256, 0, stream>>>(pos_skip, batch_skip, out_tail);
}
